// NeuronGPTOSSBlock_60936995996234
// MI455X (gfx1250) — matrix operand order repaired and checked by reading, not compiled, not run on hardware
//
#include <hip/hip_runtime.h>

// ---------------- constants (match reference) ----------------
#define TT    512
#define DD    2880
#define HH    64
#define KVH   8
#define HDIM  64
#define GG    8
#define EE    8
#define TOPK  2
#define II    2880
#define SWIN  128
#define LIMITV 7.0f
#define ALPHAV 1.702f
#define SM_SCALE 0.125f   // 1/sqrt(64)

typedef unsigned int uint;
typedef unsigned short ushort;
typedef __attribute__((ext_vector_type(16))) __bf16 bf16x16;
typedef __attribute__((ext_vector_type(8)))  float  v8f;

union Frag { bf16x16 bf; uint4 q[2]; };

// Native f32 -> bf16 (round-to-nearest-even via hardware cvt path)
static __device__ __forceinline__ ushort f2bf(float f) {
    union { __bf16 b; ushort u; } c;
    c.b = (__bf16)f;
    return c.u;
}
static __device__ __forceinline__ float bf2f(ushort h) {
    union { uint u; float f; } c; c.u = ((uint)h) << 16;
    return c.f;
}
static __device__ __forceinline__ v8f wmma_bf16(bf16x16 a, bf16x16 b, v8f c) {
    return __builtin_amdgcn_wmma_f32_16x16x32_bf16(false, a, false, b, (short)0, c, false, false);
}
static __device__ __forceinline__ v8f zero8() {
    v8f z = {0.f,0.f,0.f,0.f,0.f,0.f,0.f,0.f};
    return z;
}

// ---------------- CDNA5 async global->LDS copy (ASYNCcnt path), guarded -------------
#if defined(__AMDGCN__) && __has_builtin(__builtin_amdgcn_global_load_async_to_lds_b128) && __has_builtin(__builtin_amdgcn_s_wait_asynccnt)
#define HAVE_ASYNC_LDS 1
#endif

typedef int v4i __attribute__((vector_size(16)));
typedef __attribute__((address_space(1))) v4i* gptr_b128;
typedef __attribute__((address_space(3))) v4i* lptr_b128;

static __device__ __forceinline__ void async_copy_b128(const ushort* gsrc, ushort* ldst) {
#ifdef HAVE_ASYNC_LDS
    __builtin_amdgcn_global_load_async_to_lds_b128(
        (gptr_b128)gsrc, (lptr_b128)ldst, 0, 0);
#else
    *(uint4*)ldst = *(const uint4*)gsrc;
#endif
}
static __device__ __forceinline__ void async_copy_wait() {
#ifdef HAVE_ASYNC_LDS
    __builtin_amdgcn_s_wait_asynccnt(0);
#endif
}

// ---------------- RMSNorm (optionally with residual), writes bf16 (and fp32 residual sum) ----
__global__ __launch_bounds__(256) void rmsnorm_kernel(
    const float* __restrict__ x, const float* __restrict__ res,
    const float* __restrict__ w, float* __restrict__ xout,
    ushort* __restrict__ hbf)
{
    int t = blockIdx.x, tid = threadIdx.x;
    const float* xr = x + (size_t)t * DD;
    const float* rr = res ? res + (size_t)t * DD : nullptr;
    float ss = 0.f;
    for (int i = tid; i < DD; i += 256) {
        float v = xr[i] + (rr ? rr[i] : 0.f);
        ss += v * v;
    }
    __shared__ float red[256];
    red[tid] = ss; __syncthreads();
    for (int s = 128; s > 0; s >>= 1) {
        if (tid < s) red[tid] += red[tid + s];
        __syncthreads();
    }
    float inv = rsqrtf(red[0] / (float)DD + 1e-5f);
    for (int i = tid; i < DD; i += 256) {
        float v = xr[i] + (rr ? rr[i] : 0.f);
        if (xout) xout[(size_t)t * DD + i] = v;
        hbf[(size_t)t * DD + i] = f2bf(v * inv * w[i]);
    }
}

// ---------------- dense GEMM: C[M,N] = A_bf16[M,K] @ B_f32[K,N] + bias ------------
// block = 256 threads = 8 waves; tile M=128, N=64, K-step 32. grid = (N/64, M/128)
__global__ __launch_bounds__(256) void gemm_bf16_kernel(
    const ushort* __restrict__ A, int lda,
    const float* __restrict__ Bm, int ldb,
    const float* __restrict__ bias,
    float* __restrict__ C, int ldc, int K)
{
    __shared__ __align__(16) ushort As[128][32];
    __shared__ __align__(16) ushort Bs[64][40];
    int tid = threadIdx.x, lane = tid & 31, wave = tid >> 5;
    int m0 = blockIdx.y * 128, n0 = blockIdx.x * 64;
    v8f acc[4] = { zero8(), zero8(), zero8(), zero8() };
    int aoff = (lane >> 4) * 16;
    for (int k0 = 0; k0 < K; k0 += 32) {
        { // A tile 128x32 bf16 (async direct-to-LDS)
            int r = tid >> 1, c = (tid & 1) * 16;
            const ushort* src = A + (size_t)(m0 + r) * lda + k0 + c;
            async_copy_b128(src,     &As[r][c]);
            async_copy_b128(src + 8, &As[r][c + 8]);
        }
        { // B tile 32x64 f32 -> transposed bf16
            #pragma unroll
            for (int i = 0; i < 2; i++) {
                int lin = tid + i * 256;          // float4 index, 512 total
                int kr = lin >> 4, c4 = lin & 15;
                const float* bp = Bm + (size_t)(k0 + kr) * ldb + n0 + c4 * 4;
                if (k0 + 32 < K) __builtin_prefetch(bp + (size_t)32 * ldb, 0, 1);
                float4 f = *(const float4*)bp;
                int nn = c4 * 4;
                Bs[nn + 0][kr] = f2bf(f.x); Bs[nn + 1][kr] = f2bf(f.y);
                Bs[nn + 2][kr] = f2bf(f.z); Bs[nn + 3][kr] = f2bf(f.w);
            }
        }
        async_copy_wait();
        __syncthreads();
        Frag a;
        a.q[0] = *(const uint4*)&As[wave * 16 + (lane & 15)][aoff];
        a.q[1] = *(const uint4*)&As[wave * 16 + (lane & 15)][aoff + 8];
        #pragma unroll
        for (int j = 0; j < 4; j++) {
            Frag b;
            int n = j * 16 + (lane & 15);
            int kk = (lane >> 4) * 16;
            b.q[0] = *(const uint4*)&Bs[n][kk];
            b.q[1] = *(const uint4*)&Bs[n][kk + 8];
            acc[j] = wmma_bf16(a.bf, b.bf, acc[j]);
        }
        __syncthreads();
    }
    int rbase = m0 + wave * 16 + (lane >> 4) * 8;
    int cn = lane & 15;
    #pragma unroll
    for (int j = 0; j < 4; j++) {
        int col = n0 + j * 16 + cn;
        float bv = bias ? bias[col] : 0.f;
        #pragma unroll
        for (int r = 0; r < 8; r++)
            C[(size_t)(rbase + r) * ldc + col] = acc[j][r] + bv;
    }
}

// ---------------- YaRN RoPE: fp32 in -> bf16 out. block=32 threads (pair lanes) ------
__global__ __launch_bounds__(32) void rope_kernel(
    const float* __restrict__ in, const long long* __restrict__ pos,
    ushort* __restrict__ outb, int nheads)
{
    int t = blockIdx.x, hh = blockIdx.y, i = threadIdx.x; // i in [0,32)
    const float base = 150000.f, factor = 32.f;
    float pf = __powf(base, (float)(2 * i) / 64.f);
    float inv_extra = 1.f / pf;
    float inv_inter = 1.f / (factor * pf);
    float twolog = 2.f * logf(base);
    float lo = 64.f * logf(4096.f / (32.f * 6.2831853f)) / twolog;
    float hi = 64.f * logf(4096.f / (1.f  * 6.2831853f)) / twolog;
    lo = fmaxf(lo, 0.f); hi = fminf(hi, 63.f);
    float ramp = fminf(fmaxf(((float)i - lo) / (hi - lo), 0.f), 1.f);
    float extrap = 1.f - ramp;
    float invf = inv_inter * (1.f - extrap) + inv_extra * extrap;
    float scaling = 0.1f * logf(factor) + 1.f;
    float fr = (float)pos[t] * invf;
    float c = cosf(fr) * scaling, s = sinf(fr) * scaling;
    size_t off = (size_t)t * nheads * HDIM + (size_t)hh * HDIM;
    float x1 = in[off + i], x2 = in[off + i + 32];
    outb[off + i]      = f2bf(x1 * c - x2 * s);
    outb[off + i + 32] = f2bf(x2 * c + x1 * s);
}

// ---------------- sliding-window attention w/ sink, flash-style. 1 wave per (16 q, head) ----
__global__ __launch_bounds__(32) void attn_kernel(
    const ushort* __restrict__ Qb,   // [T][H*64] bf16
    const ushort* __restrict__ Kb,   // [T][KV*64] bf16
    const float*  __restrict__ Vf,   // [T][KV*64] f32
    const float*  __restrict__ sinks,
    ushort* __restrict__ Ob)         // [T][H*64] bf16
{
    __shared__ __align__(16) ushort Pl[16][32];
    __shared__ __align__(16) ushort Vt[64][40];
    int lane = threadIdx.x;
    int q0 = blockIdx.x * 16;
    int h = blockIdx.y;
    int kv = h >> 3;
    int hi = lane >> 4;
    int aoff = hi * 16;
    Frag aq0, aq1;
    {
        const ushort* qp = Qb + (size_t)(q0 + (lane & 15)) * (HH * HDIM) + h * HDIM;
        aq0.q[0] = *(const uint4*)(qp + aoff);
        aq0.q[1] = *(const uint4*)(qp + aoff + 8);
        aq1.q[0] = *(const uint4*)(qp + 32 + aoff);
        aq1.q[1] = *(const uint4*)(qp + 32 + aoff + 8);
    }
    float m_run[8], l_run[8];
    float sinkv = sinks[h];
    #pragma unroll
    for (int r = 0; r < 8; r++) { m_run[r] = sinkv; l_run[r] = 1.f; }
    v8f o[4] = { zero8(), zero8(), zero8(), zero8() };

    int klo = q0 - (SWIN - 1); if (klo < 0) klo = 0;
    klo &= ~31;
    int khi = q0 + 15;
    for (int kt = klo; kt <= khi; kt += 32) {
        // ---- scores: QK^T for 2 key subtiles ----
        v8f s[2];
        #pragma unroll
        for (int st = 0; st < 2; st++) {
            int keyrow = kt + st * 16 + (lane & 15);
            int keyc = keyrow < (TT - 1) ? keyrow : (TT - 1);
            const ushort* kp = Kb + (size_t)keyc * (KVH * HDIM) + kv * HDIM + hi * 16;
            Frag b0, b1;
            b0.q[0] = *(const uint4*)(kp);
            b0.q[1] = *(const uint4*)(kp + 8);
            b1.q[0] = *(const uint4*)(kp + 32);
            b1.q[1] = *(const uint4*)(kp + 40);
            v8f c = zero8();
            c = wmma_bf16(aq0.bf, b0.bf, c);
            c = wmma_bf16(aq1.bf, b1.bf, c);
            s[st] = c;
        }
        // ---- mask + online softmax ----
        int col0 = kt + (lane & 15);
        int col1 = kt + 16 + (lane & 15);
        float s0v[8], s1v[8], mx[8];
        #pragma unroll
        for (int r = 0; r < 8; r++) {
            int qrow = q0 + r + hi * 8;
            float a0 = s[0][r] * SM_SCALE;
            float a1 = s[1][r] * SM_SCALE;
            bool ok0 = (col0 <= qrow) && (col0 > qrow - SWIN);
            bool ok1 = (col1 <= qrow) && (col1 > qrow - SWIN);
            s0v[r] = ok0 ? a0 : -1e30f;
            s1v[r] = ok1 ? a1 : -1e30f;
            mx[r] = fmaxf(s0v[r], s1v[r]);
        }
        #pragma unroll
        for (int off = 1; off < 16; off <<= 1)
            #pragma unroll
            for (int r = 0; r < 8; r++)
                mx[r] = fmaxf(mx[r], __shfl_xor(mx[r], off, 32));
        float scl[8], sm[8];
        #pragma unroll
        for (int r = 0; r < 8; r++) {
            float mnew = fmaxf(m_run[r], mx[r]);
            scl[r] = __expf(m_run[r] - mnew);
            m_run[r] = mnew;
            s0v[r] = __expf(s0v[r] - mnew);
            s1v[r] = __expf(s1v[r] - mnew);
            sm[r] = s0v[r] + s1v[r];
        }
        #pragma unroll
        for (int off = 1; off < 16; off <<= 1)
            #pragma unroll
            for (int r = 0; r < 8; r++)
                sm[r] += __shfl_xor(sm[r], off, 32);
        #pragma unroll
        for (int r = 0; r < 8; r++)
            l_run[r] = l_run[r] * scl[r] + sm[r];
        #pragma unroll
        for (int j = 0; j < 4; j++)
            #pragma unroll
            for (int r = 0; r < 8; r++)
                o[j][r] *= scl[r];
        // ---- stage P (bf16) and V^T (bf16) in LDS ----
        #pragma unroll
        for (int r = 0; r < 8; r++) {
            Pl[r + hi * 8][lane & 15]        = f2bf(s0v[r]);
            Pl[r + hi * 8][16 + (lane & 15)] = f2bf(s1v[r]);
        }
        {
            int krow = kt + lane;
            int krc = krow < (TT - 1) ? krow : (TT - 1);
            const float* vp = Vf + (size_t)krc * (KVH * HDIM) + kv * HDIM;
            #pragma unroll
            for (int c4 = 0; c4 < 16; c4++) {
                float4 f = *(const float4*)(vp + c4 * 4);
                Vt[c4 * 4 + 0][lane] = f2bf(f.x);
                Vt[c4 * 4 + 1][lane] = f2bf(f.y);
                Vt[c4 * 4 + 2][lane] = f2bf(f.z);
                Vt[c4 * 4 + 3][lane] = f2bf(f.w);
            }
        }
        __syncthreads();
        // ---- PV ----
        Frag ap;
        ap.q[0] = *(const uint4*)&Pl[lane & 15][aoff];
        ap.q[1] = *(const uint4*)&Pl[lane & 15][aoff + 8];
        #pragma unroll
        for (int j = 0; j < 4; j++) {
            Frag bv;
            int n = j * 16 + (lane & 15);
            int kk = hi * 16;
            bv.q[0] = *(const uint4*)&Vt[n][kk];
            bv.q[1] = *(const uint4*)&Vt[n][kk + 8];
            o[j] = wmma_bf16(ap.bf, bv.bf, o[j]);
        }
        __syncthreads();
    }
    // ---- epilogue ----
    int cn = lane & 15;
    #pragma unroll
    for (int j = 0; j < 4; j++)
        #pragma unroll
        for (int r = 0; r < 8; r++) {
            int row = q0 + r + hi * 8;
            float val = o[j][r] / l_run[r];
            Ob[(size_t)row * (HH * HDIM) + h * HDIM + j * 16 + cn] = f2bf(val);
        }
}

// ---------------- router: logits, top-2, softmax, per-expert token lists -------------
__global__ __launch_bounds__(32) void zero_counts_kernel(int* counts) {
    if (threadIdx.x < EE) counts[threadIdx.x] = 0;
}

__global__ __launch_bounds__(256) void router_kernel(
    const ushort* __restrict__ h2bf, const float* __restrict__ rw,
    const float* __restrict__ rb, int* __restrict__ counts,
    int* __restrict__ lists, float* __restrict__ probs)
{
    int t = blockIdx.x, tid = threadIdx.x;
    float loc[EE];
    #pragma unroll
    for (int e = 0; e < EE; e++) loc[e] = 0.f;
    for (int i = tid; i < DD; i += 256) {
        float hv = bf2f(h2bf[(size_t)t * DD + i]);
        #pragma unroll
        for (int e = 0; e < EE; e++) loc[e] += hv * rw[(size_t)i * EE + e];
    }
    __shared__ float sred[256][EE];
    #pragma unroll
    for (int e = 0; e < EE; e++) sred[tid][e] = loc[e];
    __syncthreads();
    for (int s = 128; s > 0; s >>= 1) {
        if (tid < s)
            #pragma unroll
            for (int e = 0; e < EE; e++) sred[tid][e] += sred[tid + s][e];
        __syncthreads();
    }
    if (tid == 0) {
        float lg[EE];
        #pragma unroll
        for (int e = 0; e < EE; e++) lg[e] = sred[0][e] + rb[e];
        int i0 = 0;
        for (int e = 1; e < EE; e++) if (lg[e] > lg[i0]) i0 = e;
        int i1 = (i0 == 0) ? 1 : 0;
        for (int e = 0; e < EE; e++) if (e != i0 && lg[e] > lg[i1]) i1 = e;
        float e1 = __expf(lg[i1] - lg[i0]);
        float p0 = 1.f / (1.f + e1);
        float p1 = e1 / (1.f + e1);
        int pos0 = atomicAdd(&counts[i0], 1);
        int pos1 = atomicAdd(&counts[i1], 1);
        lists[i0 * TT + pos0] = t * 2 + 0;
        lists[i1 * TT + pos1] = t * 2 + 1;
        probs[t * 2 + 0] = p0;
        probs[t * 2 + 1] = p1;
    }
}

// ---------------- MoE gate_up GEMM (gathered A) + fused clamp/glu act, writes bf16 ---
__global__ __launch_bounds__(256) void moe_gateup_kernel(
    const ushort* __restrict__ h2bf,    // [T][D] bf16
    const float* __restrict__ gup,      // [E][D][2I]
    const float* __restrict__ gub,      // [E][2I]
    const int* __restrict__ counts, const int* __restrict__ lists,
    ushort* __restrict__ actbf)         // [E*512][I] bf16
{
    int e = blockIdx.z;
    int cnt = counts[e];
    int m0 = blockIdx.y * 128;
    if (m0 >= cnt) return;
    int n0 = blockIdx.x * 64;
    __shared__ __align__(16) ushort As[128][32];
    __shared__ __align__(16) ushort Bg[64][40];
    __shared__ __align__(16) ushort Bu[64][40];
    __shared__ int toks[128];
    int tid = threadIdx.x, lane = tid & 31, wave = tid >> 5;
    if (tid < 128) {
        int mi = m0 + tid;
        toks[tid] = (mi < cnt) ? (lists[e * TT + mi] >> 1) : 0;
    }
    __syncthreads();
    v8f ag[4] = { zero8(), zero8(), zero8(), zero8() };
    v8f au[4] = { zero8(), zero8(), zero8(), zero8() };
    const float* Bbase = gup + (size_t)e * DD * (2 * II);
    int aoff = (lane >> 4) * 16;
    for (int k0 = 0; k0 < DD; k0 += 32) {
        {
            int r = tid >> 1, c = (tid & 1) * 16;
            const ushort* src = h2bf + (size_t)toks[r] * DD + k0 + c;
            async_copy_b128(src,     &As[r][c]);
            async_copy_b128(src + 8, &As[r][c + 8]);
        }
        #pragma unroll
        for (int i = 0; i < 2; i++) {
            int lin = tid + i * 256;
            int kr = lin >> 4, c4 = lin & 15;
            const float* row = Bbase + (size_t)(k0 + kr) * (2 * II);
            if (k0 + 32 < DD) __builtin_prefetch(row + (size_t)32 * (2 * II) + n0 + c4 * 4, 0, 1);
            float4 fg = *(const float4*)(row + n0 + c4 * 4);
            float4 fu = *(const float4*)(row + II + n0 + c4 * 4);
            int nn = c4 * 4;
            Bg[nn + 0][kr] = f2bf(fg.x); Bg[nn + 1][kr] = f2bf(fg.y);
            Bg[nn + 2][kr] = f2bf(fg.z); Bg[nn + 3][kr] = f2bf(fg.w);
            Bu[nn + 0][kr] = f2bf(fu.x); Bu[nn + 1][kr] = f2bf(fu.y);
            Bu[nn + 2][kr] = f2bf(fu.z); Bu[nn + 3][kr] = f2bf(fu.w);
        }
        async_copy_wait();
        __syncthreads();
        Frag a;
        a.q[0] = *(const uint4*)&As[wave * 16 + (lane & 15)][aoff];
        a.q[1] = *(const uint4*)&As[wave * 16 + (lane & 15)][aoff + 8];
        #pragma unroll
        for (int j = 0; j < 4; j++) {
            Frag bg, bu;
            int n = j * 16 + (lane & 15);
            int kk = (lane >> 4) * 16;
            bg.q[0] = *(const uint4*)&Bg[n][kk]; bg.q[1] = *(const uint4*)&Bg[n][kk + 8];
            bu.q[0] = *(const uint4*)&Bu[n][kk]; bu.q[1] = *(const uint4*)&Bu[n][kk + 8];
            ag[j] = wmma_bf16(a.bf, bg.bf, ag[j]);
            au[j] = wmma_bf16(a.bf, bu.bf, au[j]);
        }
        __syncthreads();
    }
    int rloc = wave * 16 + (lane >> 4) * 8;
    int cn = lane & 15;
    #pragma unroll
    for (int j = 0; j < 4; j++) {
        int col = n0 + j * 16 + cn;
        float bgv = gub[(size_t)e * 2 * II + col];
        float buv = gub[(size_t)e * 2 * II + II + col];
        #pragma unroll
        for (int r = 0; r < 8; r++) {
            int mi = m0 + rloc + r;
            if (mi < cnt) {
                float g = fminf(ag[j][r] + bgv, LIMITV);
                float u = fminf(fmaxf(au[j][r] + buv, -LIMITV), LIMITV);
                float sg = 1.f / (1.f + __expf(-ALPHAV * g));
                float a_ = (u + 1.f) * (g * sg);
                actbf[(size_t)(e * TT + mi) * II + col] = f2bf(a_);
            }
        }
    }
}

// ---------------- MoE down GEMM + fused routing-weight scale -> slot buffer ----------
__global__ __launch_bounds__(256) void moe_down_kernel(
    const ushort* __restrict__ actbf,   // [E*512][I] bf16
    const float* __restrict__ dw,       // [E][I][D]
    const float* __restrict__ db,       // [E][D]
    const int* __restrict__ counts, const int* __restrict__ lists,
    const float* __restrict__ probs,
    float* __restrict__ slotbuf)        // [T*2][D] f32
{
    int e = blockIdx.z;
    int cnt = counts[e];
    int m0 = blockIdx.y * 128;
    if (m0 >= cnt) return;
    int n0 = blockIdx.x * 64;
    __shared__ __align__(16) ushort As[128][32];
    __shared__ __align__(16) ushort Bs[64][40];
    __shared__ int ent[128];
    int tid = threadIdx.x, lane = tid & 31, wave = tid >> 5;
    if (tid < 128) {
        int mi = m0 + tid;
        ent[tid] = (mi < cnt) ? lists[e * TT + mi] : 0;
    }
    __syncthreads();
    v8f acc[4] = { zero8(), zero8(), zero8(), zero8() };
    const float* Bbase = dw + (size_t)e * II * DD;
    const ushort* Abase = actbf + (size_t)(e * TT + m0) * II;
    int aoff = (lane >> 4) * 16;
    for (int k0 = 0; k0 < II; k0 += 32) {
        {
            int r = tid >> 1, c = (tid & 1) * 16;
            const ushort* src = Abase + (size_t)r * II + k0 + c;
            async_copy_b128(src,     &As[r][c]);
            async_copy_b128(src + 8, &As[r][c + 8]);
        }
        #pragma unroll
        for (int i = 0; i < 2; i++) {
            int lin = tid + i * 256;
            int kr = lin >> 4, c4 = lin & 15;
            const float* bp = Bbase + (size_t)(k0 + kr) * DD + n0 + c4 * 4;
            if (k0 + 32 < II) __builtin_prefetch(bp + (size_t)32 * DD, 0, 1);
            float4 f = *(const float4*)bp;
            int nn = c4 * 4;
            Bs[nn + 0][kr] = f2bf(f.x); Bs[nn + 1][kr] = f2bf(f.y);
            Bs[nn + 2][kr] = f2bf(f.z); Bs[nn + 3][kr] = f2bf(f.w);
        }
        async_copy_wait();
        __syncthreads();
        Frag a;
        a.q[0] = *(const uint4*)&As[wave * 16 + (lane & 15)][aoff];
        a.q[1] = *(const uint4*)&As[wave * 16 + (lane & 15)][aoff + 8];
        #pragma unroll
        for (int j = 0; j < 4; j++) {
            Frag b;
            int n = j * 16 + (lane & 15);
            int kk = (lane >> 4) * 16;
            b.q[0] = *(const uint4*)&Bs[n][kk];
            b.q[1] = *(const uint4*)&Bs[n][kk + 8];
            acc[j] = wmma_bf16(a.bf, b.bf, acc[j]);
        }
        __syncthreads();
    }
    int rloc = wave * 16 + (lane >> 4) * 8;
    int cn = lane & 15;
    #pragma unroll
    for (int j = 0; j < 4; j++) {
        int col = n0 + j * 16 + cn;
        float bv = db[(size_t)e * DD + col];
        #pragma unroll
        for (int r = 0; r < 8; r++) {
            int mi = m0 + rloc + r;
            if (mi < cnt) {
                int entry = ent[rloc + r];
                float p = probs[entry];
                slotbuf[(size_t)entry * DD + col] = (acc[j][r] + bv) * p;
            }
        }
    }
}

// ---------------- final combine: out = x + moe_slot0 + moe_slot1 --------------------
__global__ __launch_bounds__(256) void combine_kernel(
    const float* __restrict__ xbuf, const float* __restrict__ slotbuf,
    float* __restrict__ out)
{
    int t = blockIdx.x;
    for (int i = threadIdx.x; i < DD; i += 256) {
        float v = xbuf[(size_t)t * DD + i]
                + slotbuf[(size_t)(2 * t) * DD + i]
                + slotbuf[(size_t)(2 * t + 1) * DD + i];
        out[(size_t)t * DD + i] = v;
    }
}

// ---------------- host launcher ------------------------------------------------------
extern "C" void kernel_launch(void* const* d_in, const int* in_sizes, int n_in,
                              void* d_out, int out_size, void* d_ws, size_t ws_size,
                              hipStream_t stream) {
    const float*     hidden = (const float*)d_in[0];
    const long long* pos    = (const long long*)d_in[1];
    const float* ln1   = (const float*)d_in[2];
    const float* wq    = (const float*)d_in[3];
    const float* bq    = (const float*)d_in[4];
    const float* wk    = (const float*)d_in[5];
    const float* bk    = (const float*)d_in[6];
    const float* wv    = (const float*)d_in[7];
    const float* bv    = (const float*)d_in[8];
    const float* wo    = (const float*)d_in[9];
    const float* bo    = (const float*)d_in[10];
    const float* sinks = (const float*)d_in[11];
    const float* ln2   = (const float*)d_in[12];
    const float* rw    = (const float*)d_in[13];
    const float* rb    = (const float*)d_in[14];
    const float* gup   = (const float*)d_in[15];
    const float* gub   = (const float*)d_in[16];
    const float* dwp   = (const float*)d_in[17];
    const float* dbp   = (const float*)d_in[18];

    char* wsb = (char*)d_ws;
    size_t off = 0;
    auto take = [&](size_t bytes) -> void* {
        off = (off + 255) & ~(size_t)255;
        void* p = wsb + off;
        off += bytes;
        return p;
    };
    ushort* h_bf    = (ushort*)take((size_t)TT * DD * 2);
    float*  q_pre   = (float*) take((size_t)TT * HH * HDIM * 4);
    float*  k_pre   = (float*) take((size_t)TT * KVH * HDIM * 4);
    float*  v_pre   = (float*) take((size_t)TT * KVH * HDIM * 4);
    ushort* q_bf    = (ushort*)take((size_t)TT * HH * HDIM * 2);
    ushort* k_bf    = (ushort*)take((size_t)TT * KVH * HDIM * 2);
    ushort* attn_bf = (ushort*)take((size_t)TT * HH * HDIM * 2);
    float*  attn_o  = (float*) take((size_t)TT * DD * 4);
    float*  xbuf    = (float*) take((size_t)TT * DD * 4);
    ushort* h2_bf   = (ushort*)take((size_t)TT * DD * 2);
    int*    counts  = (int*)   take(EE * 4);
    int*    lists   = (int*)   take((size_t)EE * TT * 4);
    float*  probs   = (float*) take((size_t)TT * 2 * 4);
    ushort* act_bf  = (ushort*)take((size_t)EE * TT * II * 2);
    float*  slotbuf = (float*) take((size_t)TT * 2 * DD * 4);

    // 1. RMSNorm 1 -> h bf16
    rmsnorm_kernel<<<TT, 256, 0, stream>>>(hidden, nullptr, ln1, nullptr, h_bf);
    // 2. QKV projections
    gemm_bf16_kernel<<<dim3((HH*HDIM)/64, TT/128), 256, 0, stream>>>(
        h_bf, DD, wq, HH*HDIM, bq, q_pre, HH*HDIM, DD);
    gemm_bf16_kernel<<<dim3((KVH*HDIM)/64, TT/128), 256, 0, stream>>>(
        h_bf, DD, wk, KVH*HDIM, bk, k_pre, KVH*HDIM, DD);
    gemm_bf16_kernel<<<dim3((KVH*HDIM)/64, TT/128), 256, 0, stream>>>(
        h_bf, DD, wv, KVH*HDIM, bv, v_pre, KVH*HDIM, DD);
    // 3. RoPE -> bf16 Q/K
    rope_kernel<<<dim3(TT, HH), 32, 0, stream>>>(q_pre, pos, q_bf, HH);
    rope_kernel<<<dim3(TT, KVH), 32, 0, stream>>>(k_pre, pos, k_bf, KVH);
    // 4. attention
    attn_kernel<<<dim3(TT/16, HH), 32, 0, stream>>>(q_bf, k_bf, v_pre, sinks, attn_bf);
    // 5. O projection
    gemm_bf16_kernel<<<dim3(DD/64, TT/128), 256, 0, stream>>>(
        attn_bf, HH*HDIM, wo, DD, bo, attn_o, DD, HH*HDIM);
    // 6. residual + RMSNorm 2
    rmsnorm_kernel<<<TT, 256, 0, stream>>>(hidden, attn_o, ln2, xbuf, h2_bf);
    // 7. router
    zero_counts_kernel<<<1, 32, 0, stream>>>(counts);
    router_kernel<<<TT, 256, 0, stream>>>(h2_bf, rw, rb, counts, lists, probs);
    // 8. MoE gate_up (gathered) + activation
    moe_gateup_kernel<<<dim3(II/64, TT/128, EE), 256, 0, stream>>>(
        h2_bf, gup, gub, counts, lists, act_bf);
    // 9. MoE down (gathered) + routing-weight scale
    moe_down_kernel<<<dim3(DD/64, TT/128, EE), 256, 0, stream>>>(
        act_bf, dwp, dbp, counts, lists, probs, slotbuf);
    // 10. combine
    combine_kernel<<<TT, 256, 0, stream>>>(xbuf, slotbuf, (float*)d_out);
    (void)in_sizes; (void)n_in; (void)out_size; (void)ws_size;
}
